// SCReGAT_45827301048366
// MI455X (gfx1250) — compile-verified, matrix operand order checked by reading, not run on hardware
//
#include <hip/hip_runtime.h>
#include <math.h>

// ---------------------------------------------------------------------------
// SCReGAT for MI455X (gfx1250, wave32, WMMA) — round 2
// Invariant: NN % 16 == 0 (50000 = 3125*16), so WMMA row tiles need no guards.
// ---------------------------------------------------------------------------
#define NN   50000
#define EE1  800000
#define EE2  200000
#define GG   2568

typedef __attribute__((ext_vector_type(16))) _Float16 v16h;
typedef __attribute__((ext_vector_type(8)))  float    v8f;

// ---------------- helpers ----------------
__device__ __forceinline__ float wave_sum32(float v) {
    for (int off = 16; off >= 1; off >>= 1) v += __shfl_xor(v, off, 32);
    return v;
}
__device__ __forceinline__ float lrelu01(float x) { return x > 0.f ? x : 0.01f * x; }

// monotonic float <-> uint mapping for atomicMax on floats
__device__ __forceinline__ unsigned fmap(float f) {
    unsigned u = __float_as_uint(f);
    return (u & 0x80000000u) ? ~u : (u | 0x80000000u);
}
__device__ __forceinline__ float funmap(unsigned u) {
    return (u & 0x80000000u) ? __uint_as_float(u & 0x7FFFFFFFu) : __uint_as_float(~u);
}

// ISA 16-bit fragment K index for element e (0..15) and half-wave hi (0/1):
//   VGPR v holds K pairs; lanes 0-15 cover K 0-7/16-23, lanes 16-31 K 8-15/24-31
__device__ __forceinline__ int fragK(int e, int hi) {
    return ((e >> 3) << 4) + hi * 8 + (e & 7);
}

// ---------------------------------------------------------------------------
// 1. Node MLP:  raw_x(N,1) -> 128 (LN,lrelu) -> 64 (LN,lrelu) -> 8
//    one wave32 per node, LN via cross-lane shuffles, intermediates in LDS
// ---------------------------------------------------------------------------
__global__ __launch_bounds__(256)
void node_mlp_k(const float* __restrict__ raw_x,
                const float* __restrict__ nW1, const float* __restrict__ nb1,
                const float* __restrict__ ng1, const float* __restrict__ ngb1,
                const float* __restrict__ nW2, const float* __restrict__ nb2,
                const float* __restrict__ ng2, const float* __restrict__ ngb2,
                const float* __restrict__ nW3, const float* __restrict__ nb3,
                float* __restrict__ data, int n) {
    __shared__ float h1s[8][128];
    __shared__ float h2s[8][64];
    const int wave = threadIdx.x >> 5;
    const int lane = threadIdx.x & 31;
    const int node = blockIdx.x * 8 + wave;
    const int nc   = node < n ? node : n - 1;
    const float x  = raw_x[nc];

    float v[4]; float s = 0.f;
    #pragma unroll
    for (int i = 0; i < 4; i++) {
        int j = lane + 32 * i;
        v[i] = x * nW1[j] + nb1[j];
        s += v[i];
    }
    s = wave_sum32(s);
    float mean = s * (1.f / 128.f);
    float vs = 0.f;
    #pragma unroll
    for (int i = 0; i < 4; i++) { float d = v[i] - mean; vs += d * d; }
    vs = wave_sum32(vs);
    float inv = rsqrtf(vs * (1.f / 128.f) + 1e-5f);
    #pragma unroll
    for (int i = 0; i < 4; i++) {
        int j = lane + 32 * i;
        h1s[wave][j] = lrelu01((v[i] - mean) * inv * ng1[j] + ngb1[j]);
    }
    __syncthreads();

    float a0 = nb2[lane], a1 = nb2[lane + 32];
    for (int k = 0; k < 128; k++) {
        float hv = h1s[wave][k];
        a0 += hv * nW2[k * 64 + lane];
        a1 += hv * nW2[k * 64 + lane + 32];
    }
    float s2 = wave_sum32(a0 + a1);
    float m2 = s2 * (1.f / 64.f);
    float d0 = a0 - m2, d1 = a1 - m2;
    float vv = wave_sum32(d0 * d0 + d1 * d1);
    float inv2 = rsqrtf(vv * (1.f / 64.f) + 1e-5f);
    h2s[wave][lane]      = lrelu01(d0 * inv2 * ng2[lane]      + ngb2[lane]);
    h2s[wave][lane + 32] = lrelu01(d1 * inv2 * ng2[lane + 32] + ngb2[lane + 32]);
    __syncthreads();

    if (lane < 8 && node < n) {
        float acc = nb3[lane];
        for (int k = 0; k < 64; k++) acc += h2s[wave][k] * nW3[k * 8 + lane];
        data[node * 8 + lane] = acc;
    }
}

// ---------------------------------------------------------------------------
// 2. Edge MLP: [xs*xd, xs, xd] -> 16 (LN,lrelu) -> 8 (lrelu, sigmoid)
// ---------------------------------------------------------------------------
__global__ __launch_bounds__(256)
void edge_mlp_k(const float* __restrict__ raw_x, const int* __restrict__ ei,
                const float* __restrict__ eW1, const float* __restrict__ eb1,
                const float* __restrict__ eg1, const float* __restrict__ egb1,
                const float* __restrict__ eW2, const float* __restrict__ eb2,
                float* __restrict__ he, int E) {
    int e = blockIdx.x * blockDim.x + threadIdx.x;
    if (e >= E) return;
    float xs = raw_x[ei[e]], xd = raw_x[ei[E + e]];
    float h0 = xs * xd, h1 = xs, h2 = xd;
    float t[16]; float s = 0.f;
    #pragma unroll
    for (int j = 0; j < 16; j++) {
        t[j] = h0 * eW1[j] + h1 * eW1[16 + j] + h2 * eW1[32 + j] + eb1[j];
        s += t[j];
    }
    float mean = s * (1.f / 16.f), vs = 0.f;
    #pragma unroll
    for (int j = 0; j < 16; j++) { float d = t[j] - mean; vs += d * d; }
    float inv = rsqrtf(vs * (1.f / 16.f) + 1e-5f);
    #pragma unroll
    for (int j = 0; j < 16; j++)
        t[j] = lrelu01((t[j] - mean) * inv * eg1[j] + egb1[j]);
    #pragma unroll
    for (int o = 0; o < 8; o++) {
        float acc = eb2[o];
        #pragma unroll
        for (int j = 0; j < 16; j++) acc += t[j] * eW2[j * 8 + o];
        acc = lrelu01(acc);
        he[e * 8 + o] = 1.f / (1.f + __expf(-acc));
    }
}

// fold a_e into We:  we[h*8+k] = sum_c c1We[k,h*16+c] * c1ae[h,c]
__global__ void fold_ae_k(const float* __restrict__ We, const float* __restrict__ ae,
                          float* __restrict__ we) {
    int idx = threadIdx.x;               // 128 threads
    int h = idx >> 3, k = idx & 7;
    float s = 0.f;
    #pragma unroll
    for (int c = 0; c < 16; c++) s += We[k * 256 + h * 16 + c] * ae[h * 16 + c];
    we[h * 8 + k] = s;
}

// ---------------------------------------------------------------------------
// 3a. pack W(KIN x 256) into f16 B fragments, ISA lane order:
//     frag[(nt*32+lane)*16 + e] = W[fragK(e,hi)*256 + nt*16 + r15]  (0 if K>=KIN)
// ---------------------------------------------------------------------------
template <int KIN>
__global__ void pack_bxp_k(const float* __restrict__ W, _Float16* __restrict__ frag) {
    int idx = blockIdx.x * blockDim.x + threadIdx.x;      // 512 = 16 tiles * 32 lanes
    if (idx >= 16 * 32) return;
    int nt = idx >> 5, lane = idx & 31;
    int hi = lane >> 4, r15 = lane & 15;
    #pragma unroll
    for (int e = 0; e < 16; e++) {
        int K = fragK(e, hi);
        float v = (K < KIN) ? W[K * 256 + nt * 16 + r15] : 0.f;
        frag[(idx << 4) + e] = (_Float16)v;
    }
}

// 3b. pack W(256 x 16) into f16 B fragments per k-chunk:
//     frag[(kc*32+lane)*16 + e] = W[(kc*32+fragK(e,hi))*16 + r15]
__global__ void pack_bf_k(const float* __restrict__ W, _Float16* __restrict__ frag) {
    int idx = blockIdx.x * blockDim.x + threadIdx.x;      // 256 = 8 chunks * 32 lanes
    if (idx >= 8 * 32) return;
    int kc = idx >> 5, lane = idx & 31;
    int hi = lane >> 4, r15 = lane & 15;
    #pragma unroll
    for (int e = 0; e < 16; e++) {
        int K = kc * 32 + fragK(e, hi);
        frag[(idx << 4) + e] = (_Float16)W[K * 16 + r15];
    }
}

// ---------------------------------------------------------------------------
// 4. WMMA GEMM:  xp(N,256) = A(N,KIN) @ W(KIN,256), KIN in {8,16}, K-pad to 32
//    one wave per 16x16 tile; grid = (N/16, 4), block = 128 (4 waves)
//    A fragment: only e=0..7 can be valid (K<16); two float4 loads, no branches
// ---------------------------------------------------------------------------
template <int KIN>
__global__ __launch_bounds__(128)
void gemm_xp_k(const float* __restrict__ A, const _Float16* __restrict__ bfrag,
               float* __restrict__ xp, int n) {
    const int wave = threadIdx.x >> 5, lane = threadIdx.x & 31;
    const int hi = lane >> 4, r15 = lane & 15;
    const int m0 = blockIdx.x * 16;                 // NN % 16 == 0: no row guard
    const int nt = blockIdx.y * 4 + wave;           // output col tile, 0..15
    const int row = m0 + r15;

    v16h a = {};
    if (KIN == 16 || hi == 0) {                     // valid K range: hi*8 .. hi*8+7
        const float* ap = A + (size_t)row * KIN + hi * 8;
        float4 p0 = *(const float4*)ap;
        float4 p1 = *(const float4*)(ap + 4);
        a[0] = (_Float16)p0.x; a[1] = (_Float16)p0.y;
        a[2] = (_Float16)p0.z; a[3] = (_Float16)p0.w;
        a[4] = (_Float16)p1.x; a[5] = (_Float16)p1.y;
        a[6] = (_Float16)p1.z; a[7] = (_Float16)p1.w;
    }
    v16h b = *(const v16h*)(bfrag + (((size_t)nt * 32 + lane) << 4));

    v8f c = {};
    c = __builtin_amdgcn_wmma_f32_16x16x32_f16(false, a, false, b, (short)0, c, false, false);

    float* __restrict__ op = xp + (size_t)(m0 + hi * 8) * 256 + nt * 16 + r15;
    #pragma unroll
    for (int r = 0; r < 8; r++) op[(size_t)r * 256] = c[r];
    (void)n;
}

// ---------------------------------------------------------------------------
// 5. WMMA GEMM:  dst(N,16) = lrelu( A(N,256) @ W(256,16) + bias ), K=256
//    8 WMMA k-steps; one wave per 16-row tile; block 256 = 8 waves
// ---------------------------------------------------------------------------
__global__ __launch_bounds__(256)
void gemm_fproj_k(const float* __restrict__ A, const _Float16* __restrict__ bfrag,
                  const float* __restrict__ bias, float* __restrict__ dst, int n) {
    const int wave = threadIdx.x >> 5, lane = threadIdx.x & 31;
    const int hi = lane >> 4, r15 = lane & 15;
    const int m0 = (blockIdx.x * 8 + wave) * 16;
    if (m0 >= n) return;                            // whole-tile guard only
    const float* __restrict__ ap = A + (size_t)(m0 + r15) * 256;

    v8f c = {};
    #pragma unroll
    for (int kc = 0; kc < 8; kc++) {
        const float* q = ap + kc * 32 + hi * 8;     // K = kc*32 + hi*8 + {0..7, 16..23}
        float4 p0 = *(const float4*)q;
        float4 p1 = *(const float4*)(q + 4);
        float4 p2 = *(const float4*)(q + 16);
        float4 p3 = *(const float4*)(q + 20);
        v16h a;
        a[0]  = (_Float16)p0.x; a[1]  = (_Float16)p0.y;
        a[2]  = (_Float16)p0.z; a[3]  = (_Float16)p0.w;
        a[4]  = (_Float16)p1.x; a[5]  = (_Float16)p1.y;
        a[6]  = (_Float16)p1.z; a[7]  = (_Float16)p1.w;
        a[8]  = (_Float16)p2.x; a[9]  = (_Float16)p2.y;
        a[10] = (_Float16)p2.z; a[11] = (_Float16)p2.w;
        a[12] = (_Float16)p3.x; a[13] = (_Float16)p3.y;
        a[14] = (_Float16)p3.z; a[15] = (_Float16)p3.w;
        v16h b = *(const v16h*)(bfrag + (((size_t)kc * 32 + lane) << 4));
        c = __builtin_amdgcn_wmma_f32_16x16x32_f16(false, a, false, b, (short)0, c, false, false);
    }
    const float bv = bias[r15];
    float* __restrict__ op = dst + (size_t)(m0 + hi * 8) * 16 + r15;
    #pragma unroll
    for (int r = 0; r < 8; r++) op[(size_t)r * 16] = lrelu01(c[r] + bv);
}

// ---------------------------------------------------------------------------
// 6. per-node attention dot products, init, and edge softmax passes
// ---------------------------------------------------------------------------
__global__ __launch_bounds__(256)
void node_dots_k(const float* __restrict__ xp, const float* __restrict__ as_,
                 const float* __restrict__ ad_, float* __restrict__ sdot,
                 float* __restrict__ ddot, int n) {
    int idx = blockIdx.x * blockDim.x + threadIdx.x;
    if (idx >= n * 16) return;
    int node = idx >> 4, h = idx & 15;
    const float4* row = (const float4*)(xp + (size_t)node * 256 + h * 16);
    const float4* wa  = (const float4*)(as_ + h * 16);
    const float4* wd  = (const float4*)(ad_ + h * 16);
    float s = 0.f, d = 0.f;
    #pragma unroll
    for (int q = 0; q < 4; q++) {
        float4 v = row[q], a = wa[q], b = wd[q];
        s += v.x * a.x + v.y * a.y + v.z * a.z + v.w * a.w;
        d += v.x * b.x + v.y * b.y + v.z * b.z + v.w * b.w;
    }
    sdot[idx] = s; ddot[idx] = d;
}

__global__ __launch_bounds__(256)
void init_out_k(float* __restrict__ out, const float* __restrict__ bias, int n) {
    size_t idx = (size_t)blockIdx.x * blockDim.x + threadIdx.x;
    if (idx < (size_t)n * 256) out[idx] = bias[idx & 255];
}
__global__ __launch_bounds__(256)
void init_md_k(unsigned* __restrict__ mx, float* __restrict__ den, int n) {
    int idx = blockIdx.x * blockDim.x + threadIdx.x;
    if (idx < n * 16) { mx[idx] = 0u; den[idx] = 0.f; }
}

template <bool HAS_E>
__global__ __launch_bounds__(256)
void edge_logits_k(const int* __restrict__ ei, int E,
                   const float* __restrict__ sdot, const float* __restrict__ ddot,
                   const float* __restrict__ he, const float* __restrict__ we,
                   float* __restrict__ albuf, unsigned* __restrict__ mx) {
    size_t idx = (size_t)blockIdx.x * blockDim.x + threadIdx.x;
    if (idx >= (size_t)E * 16) return;
    int e = (int)(idx >> 4), h = (int)(idx & 15);
    int s = ei[e], d = ei[E + e];
    float al = sdot[s * 16 + h] + ddot[d * 16 + h];
    if (HAS_E) {
        const float4* hv = (const float4*)(he + (size_t)e * 8);
        const float4* w  = (const float4*)(we + h * 8);
        float4 h0 = hv[0], h1 = hv[1], w0 = w[0], w1 = w[1];
        al += h0.x * w0.x + h0.y * w0.y + h0.z * w0.z + h0.w * w0.w
            + h1.x * w1.x + h1.y * w1.y + h1.z * w1.z + h1.w * w1.w;
    }
    al = al > 0.f ? al : 0.2f * al;      // leaky 0.2
    albuf[idx] = al;
    atomicMax(&mx[d * 16 + h], fmap(al));
}

__global__ __launch_bounds__(256)
void edge_expsum_k(const int* __restrict__ ei, int E,
                   float* __restrict__ albuf, const unsigned* __restrict__ mx,
                   float* __restrict__ den) {
    size_t idx = (size_t)blockIdx.x * blockDim.x + threadIdx.x;
    if (idx >= (size_t)E * 16) return;
    int e = (int)(idx >> 4), h = (int)(idx & 15);
    int d = ei[E + e];
    float ex = __expf(albuf[idx] - funmap(mx[d * 16 + h]));
    albuf[idx] = ex;
    atomicAdd(&den[d * 16 + h], ex);
}

__global__ __launch_bounds__(256)
void edge_accum_k(const int* __restrict__ ei, int E,
                  float* __restrict__ albuf, const float* __restrict__ den,
                  const float* __restrict__ xp, float* __restrict__ out) {
    size_t idx = (size_t)blockIdx.x * blockDim.x + threadIdx.x;
    if (idx >= (size_t)E * 16) return;
    int e = (int)(idx >> 4), h = (int)(idx & 15);
    int s = ei[e], d = ei[E + e];
    float alpha = albuf[idx] / (den[d * 16 + h] + 1e-16f);
    albuf[idx] = alpha;                                   // layer-1: final atten output
    const float4* xs = (const float4*)(xp + (size_t)s * 256 + h * 16);
    float4 x0 = xs[0], x1 = xs[1], x2 = xs[2], x3 = xs[3];
    float* od = out + (size_t)d * 256 + h * 16;
    atomicAdd(&od[0],  alpha * x0.x); atomicAdd(&od[1],  alpha * x0.y);
    atomicAdd(&od[2],  alpha * x0.z); atomicAdd(&od[3],  alpha * x0.w);
    atomicAdd(&od[4],  alpha * x1.x); atomicAdd(&od[5],  alpha * x1.y);
    atomicAdd(&od[6],  alpha * x1.z); atomicAdd(&od[7],  alpha * x1.w);
    atomicAdd(&od[8],  alpha * x2.x); atomicAdd(&od[9],  alpha * x2.y);
    atomicAdd(&od[10], alpha * x2.z); atomicAdd(&od[11], alpha * x2.w);
    atomicAdd(&od[12], alpha * x3.x); atomicAdd(&od[13], alpha * x3.y);
    atomicAdd(&od[14], alpha * x3.z); atomicAdd(&od[15], alpha * x3.w);
}

// ---------------------------------------------------------------------------
// 7. genes: dat = d1 + d2; gene_out = logsumexp - g0; ct = mean
// ---------------------------------------------------------------------------
__global__ __launch_bounds__(256)
void gene_k(const float* __restrict__ d1, const float* __restrict__ d2,
            float* __restrict__ gene_out, float* __restrict__ ct, int G) {
    int i = blockIdx.x * blockDim.x + threadIdx.x;
    if (i >= G) return;
    float g[16]; float mx = -1e30f, sum = 0.f;
    #pragma unroll
    for (int j = 0; j < 16; j++) {
        g[j] = d1[i * 16 + j] + d2[i * 16 + j];
        mx = fmaxf(mx, g[j]); sum += g[j];
    }
    float se = 0.f;
    #pragma unroll
    for (int j = 0; j < 16; j++) se += __expf(g[j] - mx);
    gene_out[i] = __logf(se) + mx - g[0];
    ct[i] = sum * (1.f / 16.f);
}

// ---------------------------------------------------------------------------
// 8. cell-type classifier (single block, 128 threads)
// ---------------------------------------------------------------------------
__global__ __launch_bounds__(128)
void classifier_k(const float* __restrict__ ct, int G,
                  const float* __restrict__ cW1, const float* __restrict__ cb1,
                  const float* __restrict__ cg1, const float* __restrict__ cgb1,
                  const float* __restrict__ cW2, const float* __restrict__ cb2,
                  const float* __restrict__ cg2, const float* __restrict__ cgb2,
                  const float* __restrict__ cW3, const float* __restrict__ cb3,
                  float* __restrict__ cell) {
    __shared__ float c1[128], c2[64], red[128], lg[19];
    const int t = threadIdx.x;
    float acc = cb1[t];
    for (int i = 0; i < G; i++) acc += ct[i] * cW1[(size_t)i * 128 + t];
    red[t] = acc; __syncthreads();
    for (int off = 64; off >= 1; off >>= 1) { if (t < off) red[t] += red[t + off]; __syncthreads(); }
    float mean = red[0] * (1.f / 128.f); __syncthreads();
    float dv = acc - mean; red[t] = dv * dv; __syncthreads();
    for (int off = 64; off >= 1; off >>= 1) { if (t < off) red[t] += red[t + off]; __syncthreads(); }
    float inv = rsqrtf(red[0] * (1.f / 128.f) + 1e-5f); __syncthreads();
    c1[t] = lrelu01(dv * inv * cg1[t] + cgb1[t]); __syncthreads();

    float a2 = 0.f;
    if (t < 64) { a2 = cb2[t]; for (int k = 0; k < 128; k++) a2 += c1[k] * cW2[k * 64 + t]; }
    red[t] = (t < 64) ? a2 : 0.f; __syncthreads();
    for (int off = 64; off >= 1; off >>= 1) { if (t < off) red[t] += red[t + off]; __syncthreads(); }
    float m2 = red[0] * (1.f / 64.f); __syncthreads();
    float d2v = a2 - m2; red[t] = (t < 64) ? d2v * d2v : 0.f; __syncthreads();
    for (int off = 64; off >= 1; off >>= 1) { if (t < off) red[t] += red[t + off]; __syncthreads(); }
    float inv2 = rsqrtf(red[0] * (1.f / 64.f) + 1e-5f); __syncthreads();
    if (t < 64) c2[t] = lrelu01(d2v * inv2 * cg2[t] + cgb2[t]); __syncthreads();

    if (t < 19) { float a = cb3[t]; for (int k = 0; k < 64; k++) a += c2[k] * cW3[k * 19 + t]; lg[t] = a; }
    __syncthreads();
    if (t == 0) {
        float m = lg[0]; for (int j = 1; j < 19; j++) m = fmaxf(m, lg[j]);
        float s = 0.f;   for (int j = 0; j < 19; j++) s += __expf(lg[j] - m);
        for (int j = 0; j < 19; j++) cell[j] = __expf(lg[j] - m) / s;
    }
}

// ---------------------------------------------------------------------------
// launcher
// ---------------------------------------------------------------------------
extern "C" void kernel_launch(void* const* d_in, const int* in_sizes, int n_in,
                              void* d_out, int out_size, void* d_ws, size_t ws_size,
                              hipStream_t stream) {
    // setup_inputs() dict insertion order, params dict expanded in-place:
    const float* raw_x = (const float*)d_in[1];
    const float* nW1 = (const float*)d_in[2],  *nb1 = (const float*)d_in[3];
    const float* ng1 = (const float*)d_in[4],  *ngb1 = (const float*)d_in[5];
    const float* nW2 = (const float*)d_in[6],  *nb2 = (const float*)d_in[7];
    const float* ng2 = (const float*)d_in[8],  *ngb2 = (const float*)d_in[9];
    const float* nW3 = (const float*)d_in[10], *nb3 = (const float*)d_in[11];
    const float* eW1 = (const float*)d_in[12], *eb1 = (const float*)d_in[13];
    const float* eg1 = (const float*)d_in[14], *egb1 = (const float*)d_in[15];
    const float* eW2 = (const float*)d_in[16], *eb2 = (const float*)d_in[17];
    const float* c1W = (const float*)d_in[18], *c1as = (const float*)d_in[19];
    const float* c1ad = (const float*)d_in[20], *c1We = (const float*)d_in[21];
    const float* c1ae = (const float*)d_in[22], *c1b = (const float*)d_in[23];
    const float* f1W = (const float*)d_in[24], *f1b = (const float*)d_in[25];
    const float* c2W = (const float*)d_in[26], *c2as = (const float*)d_in[27];
    const float* c2ad = (const float*)d_in[28], *c2b = (const float*)d_in[29];
    const float* f2W = (const float*)d_in[30], *f2b = (const float*)d_in[31];
    const float* cW1 = (const float*)d_in[32], *cb1 = (const float*)d_in[33];
    const float* cg1 = (const float*)d_in[34], *cgb1 = (const float*)d_in[35];
    const float* cW2 = (const float*)d_in[36], *cb2 = (const float*)d_in[37];
    const float* cg2 = (const float*)d_in[38], *cgb2 = (const float*)d_in[39];
    const float* cW3 = (const float*)d_in[40], *cb3 = (const float*)d_in[41];
    const int* ei1 = (const int*)d_in[42];
    const int* ei2 = (const int*)d_in[43];

    // workspace layout (floats; every offset is a multiple of 8 floats = 32B)
    float* ws = (float*)d_ws;
    size_t o = 0;
    float*    w_data = ws + o; o += (size_t)NN * 8;
    float*    w_he   = ws + o; o += (size_t)EE1 * 8;
    float*    w_xp   = ws + o; o += (size_t)NN * 256;
    float*    w_s    = ws + o; o += (size_t)NN * 16;
    float*    w_d    = ws + o; o += (size_t)NN * 16;
    unsigned* w_mx   = (unsigned*)(ws + o); o += (size_t)NN * 16;
    float*    w_den  = ws + o; o += (size_t)NN * 16;
    float*    w_out  = ws + o; o += (size_t)NN * 256;
    float*    w_d1   = ws + o; o += (size_t)NN * 16;
    float*    w_d2   = ws + o; o += (size_t)NN * 16;
    float*    w_al2  = ws + o; o += (size_t)EE2 * 16;
    float*    w_we   = ws + o; o += 128;
    float*    w_ct   = ws + o; o += GG;               // 2568 (multiple of 8)
    _Float16* w_bx   = (_Float16*)(ws + o); o += (16 * 32 * 16) / 2;  // 16KB frag buf
    _Float16* w_bf   = (_Float16*)(ws + o); o += (8 * 32 * 16) / 2;   // 8KB frag buf

    // output layout: gene_out (G) | atten_w1 (E1*16) | cell_type (19)
    float* out_f   = (float*)d_out;
    float* o_gene  = out_f;
    float* o_atten = out_f + GG;                  // doubles as layer-1 alpha buffer
    float* o_cell  = out_f + GG + (size_t)EE1 * 16;

    const dim3 b256(256);

    // --- node + edge MLPs ---
    node_mlp_k<<<dim3((NN + 7) / 8), b256, 0, stream>>>(raw_x, nW1, nb1, ng1, ngb1,
                                                        nW2, nb2, ng2, ngb2, nW3, nb3, w_data, NN);
    edge_mlp_k<<<dim3((EE1 + 255) / 256), b256, 0, stream>>>(raw_x, ei1, eW1, eb1, eg1, egb1,
                                                             eW2, eb2, w_he, EE1);
    fold_ae_k<<<dim3(1), dim3(128), 0, stream>>>(c1We, c1ae, w_we);

    // --- GAT layer 1 ---
    pack_bxp_k<8><<<dim3(2), b256, 0, stream>>>(c1W, w_bx);
    pack_bf_k<<<dim3(1), b256, 0, stream>>>(f1W, w_bf);
    gemm_xp_k<8><<<dim3(NN / 16, 4), dim3(128), 0, stream>>>(w_data, w_bx, w_xp, NN);
    node_dots_k<<<dim3((NN * 16 + 255) / 256), b256, 0, stream>>>(w_xp, c1as, c1ad, w_s, w_d, NN);
    init_out_k<<<dim3((unsigned)(((size_t)NN * 256 + 255) / 256)), b256, 0, stream>>>(w_out, c1b, NN);
    init_md_k<<<dim3((NN * 16 + 255) / 256), b256, 0, stream>>>(w_mx, w_den, NN);
    {
        dim3 ge((unsigned)(((size_t)EE1 * 16 + 255) / 256));
        edge_logits_k<true><<<ge, b256, 0, stream>>>(ei1, EE1, w_s, w_d, w_he, w_we, o_atten, w_mx);
        edge_expsum_k<<<ge, b256, 0, stream>>>(ei1, EE1, o_atten, w_mx, w_den);
        edge_accum_k <<<ge, b256, 0, stream>>>(ei1, EE1, o_atten, w_den, w_xp, w_out);
    }
    gemm_fproj_k<<<dim3((NN + 127) / 128), b256, 0, stream>>>(w_out, w_bf, f1b, w_d1, NN);

    // --- GAT layer 2 (reuses xp/out/mx/den/frag buffers) ---
    pack_bxp_k<16><<<dim3(2), b256, 0, stream>>>(c2W, w_bx);
    pack_bf_k<<<dim3(1), b256, 0, stream>>>(f2W, w_bf);
    gemm_xp_k<16><<<dim3(NN / 16, 4), dim3(128), 0, stream>>>(w_d1, w_bx, w_xp, NN);
    node_dots_k<<<dim3((NN * 16 + 255) / 256), b256, 0, stream>>>(w_xp, c2as, c2ad, w_s, w_d, NN);
    init_out_k<<<dim3((unsigned)(((size_t)NN * 256 + 255) / 256)), b256, 0, stream>>>(w_out, c2b, NN);
    init_md_k<<<dim3((NN * 16 + 255) / 256), b256, 0, stream>>>(w_mx, w_den, NN);
    {
        dim3 ge((unsigned)(((size_t)EE2 * 16 + 255) / 256));
        edge_logits_k<false><<<ge, b256, 0, stream>>>(ei2, EE2, w_s, w_d, nullptr, nullptr, w_al2, w_mx);
        edge_expsum_k<<<ge, b256, 0, stream>>>(ei2, EE2, w_al2, w_mx, w_den);
        edge_accum_k <<<ge, b256, 0, stream>>>(ei2, EE2, w_al2, w_den, w_xp, w_out);
    }
    gemm_fproj_k<<<dim3((NN + 127) / 128), b256, 0, stream>>>(w_out, w_bf, f2b, w_d2, NN);

    // --- genes + classifier ---
    gene_k<<<dim3((GG + 255) / 256), b256, 0, stream>>>(w_d1, w_d2, o_gene, w_ct, GG);
    classifier_k<<<dim3(1), dim3(128), 0, stream>>>(w_ct, GG, cW1, cb1, cg1, cgb1,
                                                    cW2, cb2, cg2, cgb2, cW3, cb3, o_cell);
    (void)in_sizes; (void)n_in; (void)out_size; (void)ws_size;
}
